// SparseMLP_120259084972
// MI455X (gfx1250) — compile-verified, matrix operand order checked by reading, not run on hardware
//
#include <hip/hip_runtime.h>
#include <hip/hip_bf16.h>

// Problem constants
#define BB   16384
#define HH   16
#define DD   256
#define MM   1024
#define BM   64      // rows per workgroup
#define MC   256     // M-chunk kept in LDS

typedef __attribute__((ext_vector_type(16))) __bf16 v16bf;
typedef __attribute__((ext_vector_type(8)))  float  v8f;

union FragU { v16bf v; uint4 q[2]; };

static __device__ __forceinline__ unsigned short f2bf(float f) {
    union { float f; unsigned u; } v; v.f = f;
    unsigned r = v.u + 0x7FFFu + ((v.u >> 16) & 1u);   // round-to-nearest-even
    return (unsigned short)(r >> 16);
}

static __device__ __forceinline__ float gelu_exact(float x) {
    return 0.5f * x * (1.0f + erff(x * 0.7071067811865476f));
}

// A-fragment (16x32 bf16) from row-major bf16 LDS tile, stride = 256 halfwords.
// ISA layout: lane L: M = L%16, Khalf = L/16; VGPR0..3 hold K = 8*Khalf..+7,
// VGPR4..7 hold K = 8*Khalf+16..+23.  Two aligned 16B LDS loads per lane.
static __device__ __forceinline__ v16bf load_fragA_lds(const unsigned short* lds,
                                                       int row_base, int kt) {
    int lane = threadIdx.x & 31;
    int m = lane & 15, kh = lane >> 4;
    const unsigned short* p = lds + (row_base + m) * 256 + kt * 32 + kh * 8;
    FragU f;
    f.q[0] = *(const uint4*)p;            // K = base .. base+7
    f.q[1] = *(const uint4*)(p + 16);     // K = base+16 .. base+23
    return f.v;
}

// B-fragment: pre-swizzled in workspace so each lane's 16 bf16 are contiguous.
static __device__ __forceinline__ v16bf load_fragB_pk(const unsigned short* pk) {
    int lane = threadIdx.x & 31;
    const unsigned short* p = pk + lane * 16;
    FragU f;
    f.q[0] = *(const uint4*)p;
    f.q[1] = *(const uint4*)(p + 8);
    return f.v;
}

// ---- pre-pass: convert fp32 weights -> bf16 in WMMA B-fragment order ----
// element e of lane: e<8 -> K = kh*8 + e ; else K = kh*8 + 16 + (e-8); N = lane%16
__global__ void pack_up_kernel(const float* __restrict__ up, unsigned short* __restrict__ dst) {
    int idx  = blockIdx.x * 256 + threadIdx.x;        // H*8*64*512 = 4194304
    int e    = idx & 15;
    int lane = (idx >> 4) & 31;
    int tile = idx >> 9;
    int nt   = tile & 63;                              // M/16 tiles
    int kt   = (tile >> 6) & 7;                        // D/32 tiles
    int h    = tile >> 9;
    int n    = lane & 15, kh = lane >> 4;
    int k    = kt * 32 + kh * 8 + (e < 8 ? e : 16 + (e - 8));
    dst[idx] = f2bf(up[(h * DD + k) * MM + nt * 16 + n]);
}

__global__ void pack_down_kernel(const float* __restrict__ down, unsigned short* __restrict__ dst) {
    int idx  = blockIdx.x * 256 + threadIdx.x;        // H*32*16*512 = 4194304
    int e    = idx & 15;
    int lane = (idx >> 4) & 31;
    int tile = idx >> 9;
    int dt   = tile & 15;                              // D/16 tiles
    int kt   = (tile >> 4) & 31;                       // M/32 tiles
    int h    = tile >> 9;
    int n    = lane & 15, kh = lane >> 4;
    int k    = kt * 32 + kh * 8 + (e < 8 ? e : 16 + (e - 8));
    dst[idx] = f2bf(down[(h * MM + k) * DD + dt * 16 + n]);
}

// ---- fused per-head MLP: gather(perm) -> GEMM(up) -> GELU -> GEMM(down) -> scatter(perm) ----
// Wave tile assignment: each of the 8 waves owns ALL 4 row-tiles and 2 column
// tiles (nt/dt = wid*2 + j). No two waves load the same B fragment -> global
// (L2) traffic per WMMA is 256 B instead of 1 KB; A fragments stream from LDS.
__global__ void __launch_bounds__(256)
sparse_mlp_kernel(const float* __restrict__ x,
                  const unsigned short* __restrict__ up_pk,
                  const unsigned short* __restrict__ dn_pk,
                  const int* __restrict__ perm,
                  float* __restrict__ out) {
    extern __shared__ unsigned char smem[];
    int*            permS = (int*)smem;                                 // 1 KB
    unsigned short* ldsA  = (unsigned short*)(smem + 1024);             // 64x256 bf16, 32 KB
    unsigned short* ldsH  = (unsigned short*)(smem + 1024 + 64 * 256 * 2); // 64x256 bf16, 32 KB

    const int h    = blockIdx.x;          // head (fast dim: 16 co-resident blocks share x rows)
    const int rb   = blockIdx.y;          // row block (64 rows)
    const int tid  = threadIdx.x;
    const int lane = tid & 31;
    const int wid  = tid >> 5;            // 8 waves

    // stage this head's permutation slice
    permS[tid] = perm[h * DD + tid];
    __syncthreads();

    // gather permuted x slice into LDS as bf16: thread = one column, 64 rows
    {
        const int col = permS[tid];
        const float* xp = x + (size_t)(rb * BM) * (HH * DD) + col;
        #pragma unroll 4
        for (int row = 0; row < BM; ++row)
            ldsA[row * DD + tid] = f2bf(xp[(size_t)row * (HH * DD)]);
    }
    __syncthreads();

    // persistent output accumulators: tiles (r = 0..3, dt = wid*2 + j)
    v8f acc[8];
    #pragma unroll
    for (int t = 0; t < 8; ++t) acc[t] = (v8f)0.0f;

    const int n = lane & 15;
    const int rowoff = (lane >> 4) * 8;

    for (int mc = 0; mc < MM / MC; ++mc) {
        // ---- phase 1: hdn chunk [64, 256] = GELU(A @ up) ----
        v8f c[8];
        #pragma unroll
        for (int t = 0; t < 8; ++t) c[t] = (v8f)0.0f;

        #pragma unroll
        for (int kt = 0; kt < 8; ++kt) {             // K = D = 256
            v16bf a0 = load_fragA_lds(ldsA,  0, kt);
            v16bf a1 = load_fragA_lds(ldsA, 16, kt);
            v16bf a2 = load_fragA_lds(ldsA, 32, kt);
            v16bf a3 = load_fragA_lds(ldsA, 48, kt);
            #pragma unroll
            for (int j = 0; j < 2; ++j) {
                int ntG = mc * 16 + wid * 2 + j;     // global M-tile (unique per wave)
                v16bf b = load_fragB_pk(up_pk + ((size_t)(h * 8 + kt) * 64 + ntG) * 512);
                c[0 * 2 + j] = __builtin_amdgcn_wmma_f32_16x16x32_bf16(false, a0, false, b, (short)0, c[0 * 2 + j], false, false);
                c[1 * 2 + j] = __builtin_amdgcn_wmma_f32_16x16x32_bf16(false, a1, false, b, (short)0, c[1 * 2 + j], false, false);
                c[2 * 2 + j] = __builtin_amdgcn_wmma_f32_16x16x32_bf16(false, a2, false, b, (short)0, c[2 * 2 + j], false, false);
                c[3 * 2 + j] = __builtin_amdgcn_wmma_f32_16x16x32_bf16(false, a3, false, b, (short)0, c[3 * 2 + j], false, false);
            }
        }
        // GELU + store bf16 into ldsH
        #pragma unroll
        for (int r = 0; r < 4; ++r) {
            #pragma unroll
            for (int j = 0; j < 2; ++j) {
                int ncol = (wid * 2 + j) * 16 + n;
                #pragma unroll
                for (int i = 0; i < 8; ++i) {
                    float g = gelu_exact(c[r * 2 + j][i]);
                    ldsH[(r * 16 + rowoff + i) * MC + ncol] = f2bf(g);
                }
            }
        }
        __syncthreads();

        // ---- phase 2: acc += hdn_chunk @ down_chunk ----
        #pragma unroll
        for (int kt = 0; kt < 8; ++kt) {             // chunk K = 256
            v16bf a0 = load_fragA_lds(ldsH,  0, kt);
            v16bf a1 = load_fragA_lds(ldsH, 16, kt);
            v16bf a2 = load_fragA_lds(ldsH, 32, kt);
            v16bf a3 = load_fragA_lds(ldsH, 48, kt);
            int ktG = mc * 8 + kt;                   // global M-tile as K
            #pragma unroll
            for (int j = 0; j < 2; ++j) {
                int dt = wid * 2 + j;
                v16bf b = load_fragB_pk(dn_pk + ((size_t)(h * 32 + ktG) * 16 + dt) * 512);
                acc[0 * 2 + j] = __builtin_amdgcn_wmma_f32_16x16x32_bf16(false, a0, false, b, (short)0, acc[0 * 2 + j], false, false);
                acc[1 * 2 + j] = __builtin_amdgcn_wmma_f32_16x16x32_bf16(false, a1, false, b, (short)0, acc[1 * 2 + j], false, false);
                acc[2 * 2 + j] = __builtin_amdgcn_wmma_f32_16x16x32_bf16(false, a2, false, b, (short)0, acc[2 * 2 + j], false, false);
                acc[3 * 2 + j] = __builtin_amdgcn_wmma_f32_16x16x32_bf16(false, a3, false, b, (short)0, acc[3 * 2 + j], false, false);
            }
        }
        __syncthreads();                             // before next chunk overwrites ldsH
    }

    // ---- scatter output through perm (perm is inverse of unperm gather) ----
    #pragma unroll
    for (int r = 0; r < 4; ++r) {
        #pragma unroll
        for (int j = 0; j < 2; ++j) {
            int dt = wid * 2 + j;
            int dcol = permS[dt * 16 + n];
            #pragma unroll
            for (int i = 0; i < 8; ++i) {
                size_t row = (size_t)(rb * BM + r * 16 + rowoff + i);
                out[row * (HH * DD) + dcol] = acc[r * 2 + j][i];
            }
        }
    }
}

extern "C" void kernel_launch(void* const* d_in, const int* in_sizes, int n_in,
                              void* d_out, int out_size, void* d_ws, size_t ws_size,
                              hipStream_t stream) {
    const float* x    = (const float*)d_in[0];
    const float* up   = (const float*)d_in[1];
    const float* down = (const float*)d_in[2];
    const int*   perm = (const int*)d_in[3];
    // d_in[4] = unperm (unused; scatter via perm is its inverse)
    float* out = (float*)d_out;

    unsigned short* up_pk = (unsigned short*)d_ws;                  // 8 MB
    unsigned short* dn_pk = up_pk + (size_t)HH * DD * MM;           // 8 MB

    pack_up_kernel  <<<(HH * DD * MM) / 256, 256, 0, stream>>>(up,   up_pk);
    pack_down_kernel<<<(HH * MM * DD) / 256, 256, 0, stream>>>(down, dn_pk);

    dim3 grid(HH, BB / BM);
    size_t shmem = 1024 + 2 * (size_t)BM * MC * sizeof(unsigned short); // 66560 B
    sparse_mlp_kernel<<<grid, 256, shmem, stream>>>(x, up_pk, dn_pk, perm, out);
}
